// BaseDGCNNGFModule_19052474925313
// MI455X (gfx1250) — matrix operand order repaired and checked by reading, not compiled, not run on hardware
//
#include <hip/hip_runtime.h>
#include <hip/hip_bf16.h>

// ---------------------------------------------------------------------------
// DGCNN grouping module, fused for MI455X (gfx1250, wave32, WMMA).
//   B=8, N=4096, C=64, K=20, D=128
// Pipeline:
//   prep_weights : W1/W2/W3 fp32 -> f16   (3 x 128x128)
//   prep_points  : points fp32 -> f16, per-point squared norm
//   knn_kernel   : gram via v_wmma_f32_16x16x32_f16; column tiles staged into
//                  LDS with double-buffered global_load_async_to_lds_b128
//                  (ASYNCcnt); top-20 kept in registers (exec-masked insert)
//   mlp_kernel   : 3x (128x128) f16 WMMA MLP + ReLU + max-over-K, LDS-resident
// ---------------------------------------------------------------------------

#define BB 8
#define NN 4096
#define CC 64
#define KK 20
#define DD 128

typedef __attribute__((ext_vector_type(8)))  _Float16 v8h;
typedef __attribute__((ext_vector_type(16))) _Float16 v16h;
typedef __attribute__((ext_vector_type(8)))  float    v8f;

// ---- WMMA fragment loaders (contiguous k order; B frags use the same) ------
// A (16x32 f16): lane<16 -> row=lane,   elems 0..15 = K kb+0..15
//                lane>=16 -> row=lane-16, elems 0..15 = K kb+16..31
__device__ __forceinline__ v16h load_a_frag(const _Float16* rowp, int kbase, int lane) {
    int k0 = kbase + ((lane & 16) ? 16 : 0);
    v8h lo = *(const v8h*)(rowp + k0);
    v8h hi = *(const v8h*)(rowp + k0 + 8);
    v16h r;
#pragma unroll
    for (int i = 0; i < 8; ++i) { r[i] = lo[i]; r[8 + i] = hi[i]; }
    return r;
}

// ---------------------------------------------------------------------------
__global__ __launch_bounds__(256) void prep_weights(const float* __restrict__ W1,
                                                    const float* __restrict__ W2,
                                                    const float* __restrict__ W3,
                                                    _Float16* w1h, _Float16* w2h, _Float16* w3h) {
    int i = blockIdx.x * 256 + threadIdx.x;
    if (i < DD * DD) {
        w1h[i] = (_Float16)W1[i];
        w2h[i] = (_Float16)W2[i];
        w3h[i] = (_Float16)W3[i];
    }
}

__global__ __launch_bounds__(256) void prep_points(const float* __restrict__ pts,
                                                   _Float16* ph, float* sq) {
    int p = blockIdx.x * 256 + threadIdx.x;   // flat b*N+n
    if (p < BB * NN) {
        const float* src = pts + (size_t)p * CC;
        float s = 0.f;
#pragma unroll
        for (int c = 0; c < CC; ++c) {
            float v = src[c];
            s += v * v;
            ph[(size_t)p * CC + c] = (_Float16)v;
        }
        sq[p] = s;
    }
}

// ---------------------------------------------------------------------------
// KNN: 4 waves/WG, each wave owns 16 rows; 256 column tiles of 16 points.
// Column tiles (16 x 64 halves = 2KB) are double-buffered in LDS via the
// gfx1250 async-to-LDS engine: each of the 128 threads copies one 16B chunk.
// rank value = sq[col] - 2*gram  (sq[row] constant per row -> dropped).
__global__ __launch_bounds__(128) void knn_kernel(const _Float16* __restrict__ ph,
                                                  const float* __restrict__ sq,
                                                  int* __restrict__ knn_idx) {
    __shared__ __align__(32) _Float16 colbuf[2][16 * CC];   // 2 x 2KB
    __shared__ __align__(16) float tile[4][16][20];         // pitch 20 -> 16B rows
    const int tid  = threadIdx.x;
    const int wv   = tid >> 5;
    const int lane = tid & 31;
    const int b    = blockIdx.x / (NN / 64);
    const int m0   = (blockIdx.x % (NN / 64)) * 64 + wv * 16;

    const _Float16* pb = ph + (size_t)b * NN * CC;

    // Kick off async stage of column tile 0 (each thread: 16 bytes).
    {
        const _Float16* g = pb + (size_t)tid * 8;                       // tile 0
        unsigned l = (unsigned)(size_t)(&colbuf[0][0]) + (unsigned)tid * 16u;
        asm volatile("global_load_async_to_lds_b128 %0, %1, off"
                     :: "v"(l), "v"(g) : "memory");
    }

    // A fragments for this wave's 16 rows (K = 64 -> two 16x32 frags).
    const int mrow = m0 + (lane & 15);
    const _Float16* arow = pb + (size_t)mrow * CC;
    const v16h a0 = load_a_frag(arow, 0,  lane);
    const v16h a1 = load_a_frag(arow, 32, lane);

    float bd[KK]; int bi[KK];
#pragma unroll
    for (int t = 0; t < KK; ++t) { bd[t] = 3.4e38f; bi[t] = 0; }

    asm volatile("s_wait_asynccnt 0" ::: "memory");
    __syncthreads();

    for (int ct = 0; ct < NN / 16; ++ct) {
        // Prefetch next column tile into the other LDS buffer.
        if (ct + 1 < NN / 16) {
            const _Float16* g = pb + (size_t)(ct + 1) * 16 * CC + (size_t)tid * 8;
            unsigned l = (unsigned)(size_t)(&colbuf[(ct + 1) & 1][0]) + (unsigned)tid * 16u;
            asm volatile("global_load_async_to_lds_b128 %0, %1, off"
                         :: "v"(l), "v"(g) : "memory");
        }

        // B frags (32x16): lane<16 -> col=lane, K=kb+0..15; lane>=16 -> col=lane-16, K=kb+16..31
        const _Float16* colp = &colbuf[ct & 1][(lane & 15) * CC];
        const int k0 = (lane & 16) ? 16 : 0;
        v16h b0 = *(const v16h*)(colp + k0);
        v16h b1 = *(const v16h*)(colp + 32 + k0);
        v8f acc = {};
        acc = __builtin_amdgcn_wmma_f32_16x16x32_f16(false, a0, false, b0, (short)0, acc, false, false);
        acc = __builtin_amdgcn_wmma_f32_16x16x32_f16(false, a1, false, b1, (short)0, acc, false, false);

        const float sqn = sq[b * NN + ct * 16 + (lane & 15)];
        const int rbase = (lane & 16) ? 8 : 0;
#pragma unroll
        for (int r = 0; r < 8; ++r)
            tile[wv][rbase + r][lane & 15] = sqn - 2.0f * acc[r];

        // LDS is in-order per wave; block compiler reordering + wait stores.
        asm volatile("s_wait_dscnt 0" ::: "memory");

        if (lane < 16) {
#pragma unroll
            for (int q = 0; q < 4; ++q) {
                float4 v4 = *(const float4*)&tile[wv][lane][q * 4];
                float vv[4] = { v4.x, v4.y, v4.z, v4.w };
#pragma unroll
                for (int u = 0; u < 4; ++u) {
                    float v = vv[u];
                    if (v < bd[KK - 1]) {
                        bd[KK - 1] = v; bi[KK - 1] = ct * 16 + q * 4 + u;
#pragma unroll
                        for (int t = KK - 1; t > 0; --t) {
                            if (bd[t] < bd[t - 1]) {
                                float td = bd[t]; bd[t] = bd[t - 1]; bd[t - 1] = td;
                                int   ti = bi[t]; bi[t] = bi[t - 1]; bi[t - 1] = ti;
                            }
                        }
                    }
                }
            }
        }

        // Next tile must be resident in LDS (all waves) before anyone reads it,
        // and all reads of the current buffer done before it is re-targeted.
        asm volatile("s_wait_asynccnt 0" ::: "memory");
        __syncthreads();
    }

    if (lane < 16) {
        int* op = knn_idx + (size_t)(b * NN + mrow) * KK;
#pragma unroll
        for (int t = 0; t < KK; ++t) op[t] = bi[t];
    }
}

// ---------------------------------------------------------------------------
// One MLP layer: dst = relu(src @ W^T + bias), src/dst are 80x128 f16 in LDS.
// Wave wv owns output columns [16*wv, 16*wv+16); W-tile regs reused over 5 M-tiles.
#define ROWS 80          /* 4 points x K=20 */
#define PITCH 136        /* 128 + 8 halves: 272B rows, 16B aligned sublanes */

__device__ __forceinline__ void mlp_layer(const _Float16 (*src)[PITCH],
                                          _Float16 (*dst)[PITCH],
                                          const _Float16* __restrict__ W,
                                          const float* __restrict__ bias,
                                          int wv, int lane) {
    const int ncol = wv * 16 + (lane & 15);
    v16h bf[4];
#pragma unroll
    for (int ks = 0; ks < 4; ++ks)
        bf[ks] = *(const v16h*)(W + (size_t)ncol * DD + ks * 32 + ((lane & 16) ? 16 : 0));
    const float bv = bias[ncol];
    const int rbase = (lane & 16) ? 8 : 0;

#pragma unroll
    for (int mt = 0; mt < ROWS / 16; ++mt) {
        const _Float16* rp = &src[mt * 16 + (lane & 15)][0];
        v8f acc = {};
#pragma unroll
        for (int ks = 0; ks < 4; ++ks) {
            v16h a = load_a_frag(rp, ks * 32, lane);
            acc = __builtin_amdgcn_wmma_f32_16x16x32_f16(false, a, false, bf[ks], (short)0, acc, false, false);
        }
#pragma unroll
        for (int r = 0; r < 8; ++r) {
            float v = acc[r] + bv;
            dst[mt * 16 + rbase + r][ncol] = (_Float16)(v > 0.f ? v : 0.f);
        }
    }
}

__global__ __launch_bounds__(256) void mlp_kernel(const float* __restrict__ pts,
                                                  const int* __restrict__ knn_idx,
                                                  const _Float16* __restrict__ w1h, const float* __restrict__ b1,
                                                  const _Float16* __restrict__ w2h, const float* __restrict__ b2,
                                                  const _Float16* __restrict__ w3h, const float* __restrict__ b3,
                                                  float* __restrict__ out) {
    __shared__ _Float16 bufA[ROWS][PITCH];
    __shared__ _Float16 bufB[ROWS][PITCH];
    const int tid  = threadIdx.x;
    const int lane = tid & 31;
    const int wv   = tid >> 5;
    const int pg0  = blockIdx.x * 4;          // flat (b*N+n) of first point

    // Gather edge features: feat = [nbr - center | center]  (80 x 128 f16)
    for (int tsk = tid; tsk < ROWS * 2; tsk += 256) {
        const int row = tsk >> 1, hf = tsk & 1;
        const int pp = row / KK, k = row - pp * KK;
        const int pflat = pg0 + pp;
        const int bidx  = pflat >> 12;                         // N = 4096
        const int nb    = knn_idx[(size_t)pflat * KK + k];     // in-batch index
        const float* pc = pts + (size_t)pflat * CC;
        const float* pn = pts + (size_t)(bidx * NN + nb) * CC;
        if (hf == 0) {
#pragma unroll
            for (int c = 0; c < CC; ++c) bufA[row][c] = (_Float16)(pn[c] - pc[c]);
        } else {
#pragma unroll
            for (int c = 0; c < CC; ++c) bufA[row][CC + c] = (_Float16)pc[c];
        }
    }
    __syncthreads();

    mlp_layer(bufA, bufB, w1h, b1, wv, lane);
    __syncthreads();
    mlp_layer(bufB, bufA, w2h, b2, wv, lane);
    __syncthreads();
    mlp_layer(bufA, bufB, w3h, b3, wv, lane);
    __syncthreads();

    // max over K=20 neighbors (outputs are post-ReLU, so >= 0)
    for (int o = tid; o < 4 * DD; o += 256) {
        const int pp = o >> 7, dd = o & 127;
        float m = 0.f;
#pragma unroll
        for (int k = 0; k < KK; ++k) {
            float v = (float)bufB[pp * KK + k][dd];
            m = v > m ? v : m;
        }
        out[(size_t)(pg0 + pp) * DD + dd] = m;
    }
}

// ---------------------------------------------------------------------------
extern "C" void kernel_launch(void* const* d_in, const int* in_sizes, int n_in,
                              void* d_out, int out_size, void* d_ws, size_t ws_size,
                              hipStream_t stream) {
    const float* pts = (const float*)d_in[0];
    const float* W1  = (const float*)d_in[1];
    const float* b1  = (const float*)d_in[2];
    const float* W2  = (const float*)d_in[3];
    const float* b2  = (const float*)d_in[4];
    const float* W3  = (const float*)d_in[5];
    const float* b3  = (const float*)d_in[6];
    float* out = (float*)d_out;

    char* ws = (char*)d_ws;
    _Float16* ptsh = (_Float16*)(ws);                     // 4,194,304 B
    float*    sq   = (float*)   (ws + 4194304);           //   131,072 B
    _Float16* w1h  = (_Float16*)(ws + 4325376);           //    32,768 B
    _Float16* w2h  = (_Float16*)(ws + 4358144);           //    32,768 B
    _Float16* w3h  = (_Float16*)(ws + 4390912);           //    32,768 B
    int*      idx  = (int*)     (ws + 4423680);           // 2,621,440 B

    prep_weights<<<(DD * DD + 255) / 256, 256, 0, stream>>>(W1, W2, W3, w1h, w2h, w3h);
    prep_points <<<(BB * NN + 255) / 256, 256, 0, stream>>>(pts, ptsh, sq);
    knn_kernel  <<<BB * (NN / 64), 128, 0, stream>>>(ptsh, sq, idx);
    mlp_kernel  <<<BB * NN / 4, 256, 0, stream>>>(pts, idx, w1h, b1, w2h, b2, w3h, b3, out);
}
